// DynamicPruningGNN_84335977824257
// MI455X (gfx1250) — matrix operand order repaired and checked by reading, not compiled, not run on hardware
//
#include <hip/hip_runtime.h>
#include <math.h>

// ---------------------------------------------------------------------------
// DynamicPruningGNN on MI455X (gfx1250), wave32 + WMMA f16->f32.
// N = 4096 nodes/edges, D = 64 features, 2 layers.
// Big [N,N]x[N,64] products: WMMA with async global->LDS double-buffered B
// staging (ASYNCcnt) + register double-buffered A fragments.
// [N,N] softmax stages fused into 16-row blocks with a [16,4096] f32 strip in
// 256KB LDS (CDNA5 320KB/WGP).
// ---------------------------------------------------------------------------

#define NN 4096
#define DD 64
#define CH 64            // k-halves staged per chunk in bigmm
#define TAU_INV 1.0f
#define EPSV 1e-6f

typedef __attribute__((ext_vector_type(16))) _Float16 v16h;
typedef __attribute__((ext_vector_type(8)))  _Float16 v8h;
typedef __attribute__((ext_vector_type(8)))  float    v8f;

static __device__ __forceinline__ v8f wmma32(v16h a, v16h b, v8f c) {
  // D = A(16x32 f16) * B(32x16 f16) + C(16x16 f32)
  return __builtin_amdgcn_wmma_f32_16x16x32_f16(false, a, false, b, (short)0, c,
                                                false, false);
}

// A-fragment (16x32, M x K), row-major A with leading dim ld.
// Order used here (contiguous, not AMD's table order): lane m=lane&15, half=lane>>4; element e -> k = 16*half + e
// -> two contiguous 16B loads.
static __device__ __forceinline__ v16h load_a_frag(const _Float16* __restrict__ A,
                                                   int row0, int k0, int ld) {
  int lane = threadIdx.x & 31;
  int m = lane & 15, hf = lane >> 4;
  const _Float16* p = A + (row0 + m) * ld + k0 + hf * 16;
  v8h lo = *(const v8h*)p;
  v8h hi = *(const v8h*)(p + 8);
  v16h r;
#pragma unroll
  for (int i = 0; i < 8; ++i) { r[i] = lo[i]; r[i + 8] = hi[i]; }
  return r;
}

// B-fragment (32x16, K x N) where B[k][n] = Bt[n][k] (Bt row-major, ld).
// Order used here (contiguous, matches load_a_frag): lane n=lane&15, half=lane>>4; element e -> k = 16*half + e
// -> one contiguous 32B load per lane.
static __device__ __forceinline__ v16h load_b_fragT(const _Float16* __restrict__ Bt,
                                                    int n0, int k0, int ld) {
  int lane = threadIdx.x & 31;
  int n = lane & 15, hf = lane >> 4;
  return *(const v16h*)(Bt + (n0 + n) * ld + k0 + hf * 16);
}

// ---------------------------------------------------------------------------
// Small GEMM: out[N,64] = A[N,64](f16) @ B[64,64] (+bias) with epilogue.
// Bt[n][k] = B[k][n] (transposed f16 weights). 4 waves/block, 16-row strip/wave.
// ---------------------------------------------------------------------------
__global__ __launch_bounds__(128) void gemm64_kernel(
    const _Float16* __restrict__ A, const _Float16* __restrict__ Bt,
    const float* __restrict__ bias, const float* __restrict__ rowscale,
    float* __restrict__ outF, _Float16* __restrict__ outH,
    _Float16* __restrict__ outHT, float* __restrict__ rownorm, int relu) {
  __shared__ float tile[4][16][72];
  int wave = threadIdx.x >> 5, lane = threadIdx.x & 31;
  int row0 = (blockIdx.x * 4 + wave) * 16;
  int m = lane & 15, hf = lane >> 4;

  v16h a0 = load_a_frag(A, row0, 0, DD);
  v16h a1 = load_a_frag(A, row0, 32, DD);
#pragma unroll
  for (int nt = 0; nt < 4; ++nt) {
    v16h b0 = load_b_fragT(Bt, nt * 16, 0, DD);
    v16h b1 = load_b_fragT(Bt, nt * 16, 32, DD);
    v8f c = {};
    c = wmma32(a0, b0, c);
    c = wmma32(a1, b1, c);
#pragma unroll
    for (int r = 0; r < 8; ++r) tile[wave][r + 8 * hf][nt * 16 + m] = c[r];
  }
  __syncthreads();
  if (lane < 16) {
    int row = row0 + lane;
    float rs = rowscale ? rowscale[row] : 1.0f;
    float ss = 0.f;
    for (int c2 = 0; c2 < DD; ++c2) {
      float v = tile[wave][lane][c2];
      if (bias) v += bias[c2];
      v *= rs;
      if (relu) v = fmaxf(v, 0.f);
      ss += v * v;
      if (outF) outF[row * DD + c2] = v;
      if (outH) outH[row * DD + c2] = (_Float16)v;
      if (outHT) outHT[c2 * NN + row] = (_Float16)v;
    }
    if (rownorm) rownorm[row] = sqrtf(fmaxf(ss, 1e-16f));
  }
}

// ---------------------------------------------------------------------------
// Big-K GEMM: out[r,0:64] = sum_k A[r,k]*Bt[d,k], A:[N,N] f16 ld=N,
// Bt:[64,N] f16 ld=N.  B chunks staged global->LDS with
// GLOBAL_LOAD_ASYNC_TO_LDS_B128 (ASYNCcnt), double buffered; A fragments
// register double-buffered so WMMAs overlap both pipes.
// ---------------------------------------------------------------------------
__global__ __launch_bounds__(128) void bigmm_kernel(
    const _Float16* __restrict__ A, const _Float16* __restrict__ Bt,
    const float* __restrict__ rowscale, _Float16* __restrict__ outH,
    _Float16* __restrict__ outHT) {
  __shared__ __attribute__((aligned(32))) _Float16 Bbuf[2][64 * CH]; // 2x8KB
  __shared__ float tile[4][16][72];
  int wave = threadIdx.x >> 5, lane = threadIdx.x & 31;
  int row0 = (blockIdx.x * 4 + wave) * 16;
  int m = lane & 15, hf = lane >> 4;
  int t = threadIdx.x;

  // Stage chunk c (k = c*CH .. c*CH+63 for all 64 Bt rows) into Bbuf[buf]:
  // 512 x 16B segments, coalesced across the 128 threads, async to LDS.
  auto stage = [&](int c, int buf) {
#pragma unroll
    for (int j = 0; j < 4; ++j) {
      int seg = t + 128 * j;
      int row = seg >> 3, s8 = seg & 7;  // 8 segments per 128B row-chunk
      unsigned long long g =
          (unsigned long long)(Bt + row * NN + c * CH + s8 * 8);
      unsigned lo = (unsigned)(uintptr_t)(&Bbuf[buf][row * CH + s8 * 8]);
      asm volatile("global_load_async_to_lds_b128 %0, %1, off"
                   :: "v"(lo), "v"(g) : "memory");
    }
  };

  stage(0, 0);
  v16h a0 = load_a_frag(A, row0, 0, NN);
  v16h a1 = load_a_frag(A, row0, 32, NN);
  asm volatile("s_wait_asynccnt 0x0" ::: "memory");
  __syncthreads();

  v8f c0 = {}, c1 = {}, c2 = {}, c3 = {};
  const int NCH = NN / CH;
  for (int c = 0; c < NCH; ++c) {
    int cur = c & 1;
    if (c + 1 < NCH) stage(c + 1, cur ^ 1);      // async fill of next buffer
    v16h na0 = {}, na1 = {};
    if (c + 1 < NCH) {                           // next A frags in flight
      na0 = load_a_frag(A, row0, (c + 1) * CH, NN);
      na1 = load_a_frag(A, row0, (c + 1) * CH + 32, NN);
    }
    const _Float16* bb = &Bbuf[cur][0];
#pragma unroll
    for (int ks = 0; ks < 2; ++ks) {
      v16h aa = ks ? a1 : a0;
      v16h b0 = *(const v16h*)(bb + ( 0 + m) * CH + ks * 32 + hf * 16);
      v16h b1 = *(const v16h*)(bb + (16 + m) * CH + ks * 32 + hf * 16);
      v16h b2 = *(const v16h*)(bb + (32 + m) * CH + ks * 32 + hf * 16);
      v16h b3 = *(const v16h*)(bb + (48 + m) * CH + ks * 32 + hf * 16);
      c0 = wmma32(aa, b0, c0);
      c1 = wmma32(aa, b1, c1);
      c2 = wmma32(aa, b2, c2);
      c3 = wmma32(aa, b3, c3);
    }
    a0 = na0; a1 = na1;
    asm volatile("s_wait_asynccnt 0x0" ::: "memory");
    __syncthreads();
  }

#pragma unroll
  for (int r = 0; r < 8; ++r) {
    tile[wave][r + 8 * hf][ 0 + m] = c0[r];
    tile[wave][r + 8 * hf][16 + m] = c1[r];
    tile[wave][r + 8 * hf][32 + m] = c2[r];
    tile[wave][r + 8 * hf][48 + m] = c3[r];
  }
  __syncthreads();
  if (lane < 16) {
    int row = row0 + lane;
    float rs = rowscale ? rowscale[row] : 1.0f;
    for (int d = 0; d < DD; ++d) {
      float v = tile[wave][lane][d] * rs;
      if (outH)  outH[row * DD + d] = (_Float16)v;
      if (outHT) outHT[d * NN + row] = (_Float16)v;
    }
  }
}

// ---------------------------------------------------------------------------
// Attention row-block: scores = (Q @ ef^T)*scale*H for 16 rows, softmax over
// the 4096-wide row held in LDS, accumulate column sums into attsum (atomic).
// B fragments register double-buffered across j-tiles.
// ---------------------------------------------------------------------------
__global__ __launch_bounds__(128) void att_kernel(
    const _Float16* __restrict__ Qh, const _Float16* __restrict__ efh,
    const float* __restrict__ Hf, float* __restrict__ attsum) {
  extern __shared__ float s[];          // [16*NN] scores + 128 partials + 16 sums
  float* rowpart = s + 16 * NN;
  float* rowsum  = rowpart + 128;
  int wave = threadIdx.x >> 5, lane = threadIdx.x & 31;
  int i0 = blockIdx.x * 16;
  int m = lane & 15, hf = lane >> 4;
  const float scale = 0.125f;           // D^-0.5

  v16h a0 = load_a_frag(Qh, i0, 0, DD);
  v16h a1 = load_a_frag(Qh, i0, 32, DD);
  v16h b0 = load_b_fragT(efh, wave * 16, 0, DD);
  v16h b1 = load_b_fragT(efh, wave * 16, 32, DD);
  for (int jt = wave; jt < NN / 16; jt += 4) {
    int j0 = jt * 16;
    v16h nb0 = {}, nb1 = {};
    if (jt + 4 < NN / 16) {
      nb0 = load_b_fragT(efh, (jt + 4) * 16, 0, DD);
      nb1 = load_b_fragT(efh, (jt + 4) * 16, 32, DD);
    }
    v8f c = {};
    c = wmma32(a0, b0, c);
    c = wmma32(a1, b1, c);
#pragma unroll
    for (int r = 0; r < 8; ++r) {
      int row = r + 8 * hf;
      float hv = Hf[(i0 + row) * NN + j0 + m];
      s[row * NN + j0 + m] = c[r] * scale * hv;   // multiplicative H mask
    }
    b0 = nb0; b1 = nb1;
  }
  __syncthreads();
  {
    int r = threadIdx.x & 15, part = threadIdx.x >> 4;  // 8 partials per row
    float acc = 0.f;
    for (int c2 = part * 512; c2 < part * 512 + 512; ++c2)
      acc += expf(s[r * NN + c2]);
    rowpart[r * 8 + part] = acc;
  }
  __syncthreads();
  if (threadIdx.x < 16) {
    float acc = 0.f;
    for (int p = 0; p < 8; ++p) acc += rowpart[threadIdx.x * 8 + p];
    rowsum[threadIdx.x] = acc;
  }
  __syncthreads();
  for (int c2 = threadIdx.x; c2 < NN; c2 += 128) {
    float acc = 0.f;
    for (int r = 0; r < 16; ++r) acc += expf(s[r * NN + c2]) / rowsum[r];
    atomicAdd(&attsum[c2], acc);        // one atomic per column per block
  }
}

// ---------------------------------------------------------------------------
// Cosine row-block: cos = xn @ xn^T, scores = cos*H/TAU, row softmax in LDS,
// threshold -> H_pruned (f16 0/1) + H_pruned^T, row degrees (direct) and
// column degrees (atomic).
// ---------------------------------------------------------------------------
__global__ __launch_bounds__(128) void cos_kernel(
    const _Float16* __restrict__ xnh, const float* __restrict__ Hf,
    const float* __restrict__ cmask, _Float16* __restrict__ Hp,
    _Float16* __restrict__ HpT, float* __restrict__ conn,
    float* __restrict__ edeg, float th_edge) {
  extern __shared__ float s[];
  float* rowpart = s + 16 * NN;
  float* rowsum  = rowpart + 128;
  int wave = threadIdx.x >> 5, lane = threadIdx.x & 31;
  int i0 = blockIdx.x * 16;
  int m = lane & 15, hf = lane >> 4;

  v16h a0 = load_a_frag(xnh, i0, 0, DD);
  v16h a1 = load_a_frag(xnh, i0, 32, DD);
  v16h b0 = load_b_fragT(xnh, wave * 16, 0, DD);
  v16h b1 = load_b_fragT(xnh, wave * 16, 32, DD);
  for (int jt = wave; jt < NN / 16; jt += 4) {
    int j0 = jt * 16;
    v16h nb0 = {}, nb1 = {};
    if (jt + 4 < NN / 16) {
      nb0 = load_b_fragT(xnh, (jt + 4) * 16, 0, DD);
      nb1 = load_b_fragT(xnh, (jt + 4) * 16, 32, DD);
    }
    v8f c = {};
    c = wmma32(a0, b0, c);
    c = wmma32(a1, b1, c);
#pragma unroll
    for (int r = 0; r < 8; ++r) {
      int row = r + 8 * hf;
      float hv = Hf[(i0 + row) * NN + j0 + m];
      s[row * NN + j0 + m] = c[r] * hv * TAU_INV;
    }
    b0 = nb0; b1 = nb1;
  }
  __syncthreads();
  {
    int r = threadIdx.x & 15, part = threadIdx.x >> 4;
    float acc = 0.f;
    for (int c2 = part * 512; c2 < part * 512 + 512; ++c2)
      acc += expf(s[r * NN + c2]);
    rowpart[r * 8 + part] = acc;
  }
  __syncthreads();
  if (threadIdx.x < 16) {
    float acc = 0.f;
    for (int p = 0; p < 8; ++p) acc += rowpart[threadIdx.x * 8 + p];
    rowsum[threadIdx.x] = acc;
  }
  __syncthreads();
  {
    int r = threadIdx.x & 15, part = threadIdx.x >> 4;
    float rs = rowsum[r];
    float cm = cmask[i0 + r];
    float rc = 0.f;
    for (int c2 = part * 512; c2 < part * 512 + 512; ++c2) {
      float p  = expf(s[r * NN + c2]) / rs;
      float hv = Hf[(i0 + r) * NN + c2];
      float hp = (p >= th_edge && cm != 0.f && hv != 0.f) ? 1.f : 0.f;
      Hp[(i0 + r) * NN + c2]  = (_Float16)hp;
      HpT[c2 * NN + (i0 + r)] = (_Float16)hp;
      s[r * NN + c2] = hp;              // reuse LDS for column degree pass
      rc += hp;
    }
    rowpart[r * 8 + part] = rc;
  }
  __syncthreads();
  if (threadIdx.x < 16) {
    float acc = 0.f;
    for (int p = 0; p < 8; ++p) acc += rowpart[threadIdx.x * 8 + p];
    conn[i0 + threadIdx.x] = acc;       // vdeg == connectivity
  }
  __syncthreads();
  for (int c2 = threadIdx.x; c2 < NN; c2 += 128) {
    float acc = 0.f;
    for (int r = 0; r < 16; ++r) acc += s[r * NN + c2];
    atomicAdd(&edeg[c2], acc);
  }
}

// ---------------------------------------------------------------------------
// Single-block length-4096 softmax/threshold stages.
// ---------------------------------------------------------------------------
__global__ __launch_bounds__(512) void comp_mask_kernel(
    const float* __restrict__ structural, const float* __restrict__ attsum,
    float* __restrict__ cmask, float th) {
  __shared__ float red[512];
  int t = threadIdx.x;
  float mx = -3.4e38f;
  for (int n = t; n < NN; n += 512) {
    float v = (0.5f * structural[n] + 0.5f * attsum[n] * (1.0f / NN)) * TAU_INV;
    mx = fmaxf(mx, v);
  }
  red[t] = mx; __syncthreads();
  for (int s2 = 256; s2 > 0; s2 >>= 1) {
    if (t < s2) red[t] = fmaxf(red[t], red[t + s2]);
    __syncthreads();
  }
  mx = red[0]; __syncthreads();
  float sm = 0.f;
  for (int n = t; n < NN; n += 512) {
    float v = (0.5f * structural[n] + 0.5f * attsum[n] * (1.0f / NN)) * TAU_INV;
    sm += expf(v - mx);
  }
  red[t] = sm; __syncthreads();
  for (int s2 = 256; s2 > 0; s2 >>= 1) {
    if (t < s2) red[t] += red[t + s2];
    __syncthreads();
  }
  sm = red[0];
  for (int n = t; n < NN; n += 512) {
    float v = (0.5f * structural[n] + 0.5f * attsum[n] * (1.0f / NN)) * TAU_INV;
    cmask[n] = (expf(v - mx) / sm >= th) ? 1.f : 0.f;
  }
}

__global__ __launch_bounds__(512) void node_mask_kernel(
    const float* __restrict__ conn, const float* __restrict__ featmag,
    const float* __restrict__ edeg, const _Float16* __restrict__ xc_h,
    _Float16* __restrict__ xf_h, float* __restrict__ dv_isqrt,
    float* __restrict__ de_inv, float th) {
  __shared__ float red[512];
  __shared__ float nm[NN];
  int t = threadIdx.x;
  float mx = -3.4e38f;
  for (int n = t; n < NN; n += 512) {
    float v = conn[n] * featmag[n] * TAU_INV;
    mx = fmaxf(mx, v);
  }
  red[t] = mx; __syncthreads();
  for (int s2 = 256; s2 > 0; s2 >>= 1) {
    if (t < s2) red[t] = fmaxf(red[t], red[t + s2]);
    __syncthreads();
  }
  mx = red[0]; __syncthreads();
  float sm = 0.f;
  for (int n = t; n < NN; n += 512) {
    float v = conn[n] * featmag[n] * TAU_INV;
    sm += expf(v - mx);
  }
  red[t] = sm; __syncthreads();
  for (int s2 = 256; s2 > 0; s2 >>= 1) {
    if (t < s2) red[t] += red[t + s2];
    __syncthreads();
  }
  sm = red[0];
  for (int n = t; n < NN; n += 512) {
    float v = conn[n] * featmag[n] * TAU_INV;
    float p = expf(v - mx) / sm;
    nm[n] = (p >= th && conn[n] > 0.f) ? 1.f : 0.f;  // connected == conn>0
    dv_isqrt[n] = rsqrtf(fmaxf(conn[n], EPSV));
    de_inv[n]   = 1.f / fmaxf(edeg[n], EPSV);
  }
  __syncthreads();
  for (int i = t; i < NN * DD; i += 512) {
    int n = i >> 6;
    xf_h[i] = (nm[n] != 0.f) ? xc_h[i] : (_Float16)0.f;
  }
}

// Per-node: x_comp = x*cmask, norms, xn = x_comp/||x_comp||.
__global__ __launch_bounds__(64) void normalize_kernel(
    const float* __restrict__ x, const float* __restrict__ cmask,
    _Float16* __restrict__ xc_h, _Float16* __restrict__ xn_h,
    float* __restrict__ featmag) {
  __shared__ float red[64];
  int n = blockIdx.x;
  float cm = cmask[n];
  float v = x[n * DD + threadIdx.x] * cm;
  red[threadIdx.x] = v * v; __syncthreads();
  for (int s2 = 32; s2 > 0; s2 >>= 1) {
    if (threadIdx.x < s2) red[threadIdx.x] += red[threadIdx.x + s2];
    __syncthreads();
  }
  float nrm = sqrtf(fmaxf(red[0], 1e-16f));  // safe_norm >= 1e-8 -> max(.,1e-8) no-op
  xc_h[n * DD + threadIdx.x] = (_Float16)v;
  xn_h[n * DD + threadIdx.x] = (_Float16)(v / nrm);
  if (threadIdx.x == 0) featmag[n] = nrm;
}

// ---------------------------------------------------------------------------
// Prep / utility kernels.
// ---------------------------------------------------------------------------
__global__ void zero_kernel(float* __restrict__ p, int n) {
  int i = blockIdx.x * 256 + threadIdx.x;
  if (i < n) p[i] = 0.f;
}
__global__ void f32_to_f16_kernel(const float* __restrict__ s,
                                  _Float16* __restrict__ d, int n) {
  int i = blockIdx.x * 256 + threadIdx.x;
  if (i < n) d[i] = (_Float16)s[i];
}
// dst[n][k] = (f16) src[k][n], 64x64
__global__ __launch_bounds__(64) void transpose64_kernel(
    const float* __restrict__ src, _Float16* __restrict__ dst) {
  int n = blockIdx.x, k = threadIdx.x;
  dst[n * DD + k] = (_Float16)src[k * DD + n];
}
// HT[e][n] = (f16) H[n][e], 4096x4096, 16x16 LDS tiles for coalescing.
__global__ __launch_bounds__(256) void transposeH_kernel(
    const float* __restrict__ H, _Float16* __restrict__ HT) {
  __shared__ float t[16][17];
  int tx = threadIdx.x & 15, ty = threadIdx.x >> 4;
  int bx = blockIdx.x, by = blockIdx.y;
  t[ty][tx] = H[(by * 16 + ty) * NN + bx * 16 + tx];
  __syncthreads();
  HT[(bx * 16 + ty) * NN + (by * 16 + tx)] = (_Float16)t[tx][ty];
}

// ---------------------------------------------------------------------------
// Host orchestration.
// ---------------------------------------------------------------------------
extern "C" void kernel_launch(void* const* d_in, const int* in_sizes, int n_in,
                              void* d_out, int out_size, void* d_ws, size_t ws_size,
                              hipStream_t stream) {
  (void)in_sizes; (void)n_in; (void)out_size; (void)ws_size;
  const float* X  = (const float*)d_in[0];
  const float* W  = (const float*)d_in[1];
  const float* Th = (const float*)d_in[2];
  const float* Wq = (const float*)d_in[3];
  const float* bq = (const float*)d_in[4];
  const float* Wv = (const float*)d_in[5];
  const float* bv = (const float*)d_in[6];
  const float* H  = (const float*)d_in[7];
  float* out = (float*)d_out;

  // ---- workspace carve (~101 MB; assumes ws_size is sufficient) ----
  char* w = (char*)d_ws;
  auto carve = [&](size_t bytes) -> void* {
    void* p = (void*)w;
    w += (bytes + 255) & ~(size_t)255;
    return p;
  };
  const size_t ND = (size_t)NN * DD;
  _Float16* HhT   = (_Float16*)carve((size_t)NN * NN * 2);  // H^T f16
  _Float16* Hp    = (_Float16*)carve((size_t)NN * NN * 2);  // H_pruned f16
  _Float16* HpT   = (_Float16*)carve((size_t)NN * NN * 2);  // H_pruned^T f16
  _Float16* xh    = (_Float16*)carve(ND * 2);   // layer input, f16
  _Float16* Qh    = (_Float16*)carve(ND * 2);
  _Float16* Vt_h  = (_Float16*)carve(ND * 2);   // V^T [64,N]
  _Float16* efh   = (_Float16*)carve(ND * 2);   // edge features [N,64]
  _Float16* xc_h  = (_Float16*)carve(ND * 2);
  _Float16* xn_h  = (_Float16*)carve(ND * 2);
  _Float16* xf_h  = (_Float16*)carve(ND * 2);
  _Float16* tst   = (_Float16*)carve(ND * 2);   // (dv*transformed)^T [64,N]
  _Float16* agg_t = (_Float16*)carve(ND * 2);   // agg^T [64,N]
  _Float16* msg_h = (_Float16*)carve(ND * 2);
  _Float16* Wt    = (_Float16*)carve((size_t)DD * DD * 2);
  _Float16* Tht   = (_Float16*)carve((size_t)DD * DD * 2);
  _Float16* Wqt   = (_Float16*)carve((size_t)DD * DD * 2);
  _Float16* Wvt   = (_Float16*)carve((size_t)DD * DD * 2);
  float* xbuf       = (float*)carve(ND * 4);    // layer-1 input f32
  float* structural = (float*)carve((size_t)NN * 4);
  float* attsum     = (float*)carve((size_t)NN * 4);
  float* featmag    = (float*)carve((size_t)NN * 4);
  float* conn       = (float*)carve((size_t)NN * 4);
  float* edeg       = (float*)carve((size_t)NN * 4);
  float* cmask      = (float*)carve((size_t)NN * 4);
  float* dv_isqrt   = (float*)carve((size_t)NN * 4);
  float* de_inv     = (float*)carve((size_t)NN * 4);

  const int    SM    = 16 * NN * 4 + (128 + 16) * 4;  // 256KB + partials
  const dim3   B128(128), B256(256), B64(64), B512(512);

  // ---- one-time prep ----
  transpose64_kernel<<<DD, B64, 0, stream>>>(W,  Wt);
  transpose64_kernel<<<DD, B64, 0, stream>>>(Th, Tht);
  transpose64_kernel<<<DD, B64, 0, stream>>>(Wq, Wqt);
  transpose64_kernel<<<DD, B64, 0, stream>>>(Wv, Wvt);
  f32_to_f16_kernel<<<(int)(ND / 256), B256, 0, stream>>>(X, xh, (int)ND);
  transposeH_kernel<<<dim3(NN / 16, NN / 16), B256, 0, stream>>>(H, HhT);

  for (int it = 0; it < 2; ++it) {
    float tb = 0.000244f + (1.0f - expf(-0.1f * (float)it)) * (0.0003f - 0.000244f);
    const float th_comp = tb, th_edge = tb, th_node = tb;  // min(tb,0.8/0.7)==tb
    const float* xf32 = (it == 0) ? X : xbuf;
    float* outF = (it == 1) ? out : xbuf;

    zero_kernel<<<NN / 256, B256, 0, stream>>>(attsum, NN);
    zero_kernel<<<NN / 256, B256, 0, stream>>>(edeg, NN);

    // structural = ||x@W|| ; Q = x@Wq+bq ; V^T = (x@Wv+bv)^T
    gemm64_kernel<<<NN / 64, B128, 0, stream>>>(xh, Wt, nullptr, nullptr,
        nullptr, nullptr, nullptr, structural, 0);
    gemm64_kernel<<<NN / 64, B128, 0, stream>>>(xh, Wqt, bq, nullptr,
        nullptr, Qh, nullptr, nullptr, 0);
    gemm64_kernel<<<NN / 64, B128, 0, stream>>>(xh, Wvt, bv, nullptr,
        nullptr, nullptr, Vt_h, nullptr, 0);
    // edge_feat = H^T @ V
    bigmm_kernel<<<NN / 64, B128, 0, stream>>>(HhT, Vt_h, nullptr, efh, nullptr);
    // attention softmax + column means
    att_kernel<<<NN / 16, B128, SM, stream>>>(Qh, efh, H, attsum);
    // component mask
    comp_mask_kernel<<<1, B512, 0, stream>>>(structural, attsum, cmask, th_comp);
    // x_comp, xn, norms
    normalize_kernel<<<NN, B64, 0, stream>>>(xf32, cmask, xc_h, xn_h, featmag);
    // cosine softmax -> H_pruned, degrees
    cos_kernel<<<NN / 16, B128, SM, stream>>>(xn_h, H, cmask, Hp, HpT, conn,
                                              edeg, th_edge);
    // node mask, x_final, degree scalings
    node_mask_kernel<<<1, B512, 0, stream>>>(conn, featmag, edeg, xc_h, xf_h,
                                             dv_isqrt, de_inv, th_node);
    // t^T = (dv_isqrt * (x_final@W))^T
    gemm64_kernel<<<NN / 64, B128, 0, stream>>>(xf_h, Wt, nullptr, dv_isqrt,
        nullptr, nullptr, tst, nullptr, 0);
    // agg^T = (de_inv * (Hp^T @ t))^T
    bigmm_kernel<<<NN / 64, B128, 0, stream>>>(HpT, tst, de_inv, nullptr, agg_t);
    // msg = dv_isqrt * (Hp @ agg)
    bigmm_kernel<<<NN / 64, B128, 0, stream>>>(Hp, agg_t, dv_isqrt, msg_h, nullptr);
    // x_next = relu(msg @ Theta)  (f32 to xbuf/d_out, f16 to xh for next layer)
    gemm64_kernel<<<NN / 64, B128, 0, stream>>>(msg_h, Tht, nullptr, nullptr,
        outF, xh, nullptr, nullptr, 1);
  }
}
